// AUTOAdapterLayer_77893526880533
// MI455X (gfx1250) — matrix operand order repaired and checked by reading, not compiled, not run on hardware
//
#include <hip/hip_runtime.h>
#include <hip/hip_bf16.h>

typedef __attribute__((ext_vector_type(16))) _Float16 v16h;
typedef __attribute__((ext_vector_type(8)))  _Float16 v8h;
typedef __attribute__((ext_vector_type(8)))  float    v8f;
typedef __attribute__((ext_vector_type(4)))  float    v4f;

#define DIM_D 4096
#define DIM_R 64
#define N_TOK 16384
#define KP    128            // K-panel staged in LDS per outer step (GEMM1)
#define LDSA_STRIDE (KP + 8) // halfwords; 136*2=272B = 17*16B -> b128-aligned rows
#define LDSB_STRIDE (DIM_R + 8) // 72*2=144B = 9*16B -> b128-aligned rows

static __device__ inline v8f wmma_f16(const v16h& a, const v16h& b, const v8f& c) {
    return __builtin_amdgcn_wmma_f32_16x16x32_f16(
        /*neg_a=*/false, a, /*neg_b=*/false, b,
        /*c_mod=*/(short)0, c, /*reuse_a=*/false, /*reuse_b=*/false);
}

// ---------------------------------------------------------------------------
// Kernel 1: T[n,r] = sum_d x[n,d] * A[d,r]   (f16 out, f32 accumulate)
// Block: 128 threads = 4 waves; each wave: 16 tokens x 64 R.
// ---------------------------------------------------------------------------
__global__ __launch_bounds__(128) void lora_gemm1(
    const float* __restrict__ x, const float* __restrict__ A,
    _Float16* __restrict__ T)
{
    __shared__ _Float16 At[DIM_R * LDSA_STRIDE]; // 64*136*2 = 17408 B, transposed f16 panel

    const int tid  = threadIdx.x;
    const int wave = tid >> 5;
    const int lane = tid & 31;
    const int m    = lane & 15;   // row within tile / B column
    const int hi   = lane >> 4;   // half-wave select
    const int tok0 = blockIdx.x * 64 + wave * 16;

    const float* __restrict__ xrow = x + (size_t)(tok0 + m) * DIM_D;

    v8f c[4];
    c[0] = (v8f){}; c[1] = (v8f){}; c[2] = (v8f){}; c[3] = (v8f){};

    for (int k0 = 0; k0 < DIM_D; k0 += KP) {
        // prefetch next x panel (non-temporal-ish speculative pull into L2)
        if (k0 + KP < DIM_D)
            __builtin_prefetch(xrow + k0 + KP, 0, 1);

        __syncthreads(); // previous iteration done consuming LDS
        // Cooperative stage: lora_A[k0:k0+KP, 0:64] -> f16, transposed (r-major)
        {
            const float* __restrict__ Ap = A + (size_t)k0 * DIM_R;
            #pragma unroll 4
            for (int i = 0; i < (KP * DIM_R) / 128; ++i) { // 64 iters
                int idx = tid + 128 * i;     // coalesced over row-major A
                int kk  = idx >> 6;          // / 64
                int nn  = idx & 63;
                At[nn * LDSA_STRIDE + kk] = (_Float16)Ap[idx];
            }
        }
        __syncthreads();

        #pragma unroll
        for (int ks = 0; ks < KP; ks += 32) {
            // ---- A fragment (x tile 16x32): lane m holds K = 8*hi+0..7, 16+8*hi+0..7
            const float* p0 = xrow + k0 + ks + 8 * hi;
            const v4f f0 = *(const v4f*)(p0);
            const v4f f1 = *(const v4f*)(p0 + 4);
            const v4f f2 = *(const v4f*)(p0 + 16);
            const v4f f3 = *(const v4f*)(p0 + 20);
            v16h a;
            #pragma unroll
            for (int e = 0; e < 4; ++e) {
                a[e]      = (_Float16)f0[e];
                a[4 + e]  = (_Float16)f1[e];
                a[8 + e]  = (_Float16)f2[e];
                a[12 + e] = (_Float16)f3[e];
            }
            // ---- 4 B fragments from transposed LDS, 4 WMMAs
            #pragma unroll
            for (int nb = 0; nb < 4; ++nb) {
                // B operand: lane half hi holds K = 8*hi+0..7, 16+8*hi+0..7 (matches A)
                const v8h* bp = (const v8h*)&At[(nb * 16 + m) * LDSA_STRIDE + ks + 8 * hi];
                const v8h b0 = bp[0];
                const v8h b1 = bp[2];
                v16h b;
                #pragma unroll
                for (int e = 0; e < 8; ++e) { b[e] = b0[e]; b[8 + e] = b1[e]; }
                c[nb] = wmma_f16(a, b, c[nb]);
            }
        }
    }

    // Store T row-major [token][r] as f16 (2 MB total -> L2 resident)
    #pragma unroll
    for (int nb = 0; nb < 4; ++nb) {
        #pragma unroll
        for (int v = 0; v < 8; ++v) {
            int row = tok0 + v + 8 * hi;           // C/D layout: M = vgpr + 8*hi
            T[(size_t)row * DIM_R + nb * 16 + m] = (_Float16)c[nb][v];
        }
    }
}

// ---------------------------------------------------------------------------
// Kernel 2: out[n,d] = tw[n] * 2.0 * sum_r T[n,r] * Bw[r,d]
// Block: 256 threads = 8 waves; WG covers 128 tokens x 64 d.
// ---------------------------------------------------------------------------
__global__ __launch_bounds__(256) void lora_gemm2(
    const _Float16* __restrict__ T, const float* __restrict__ Bw,
    const float* __restrict__ tw, float* __restrict__ out)
{
    __shared__ _Float16 Bt[DIM_R * LDSB_STRIDE]; // 64*72*2 = 9216 B, d-major f16 panel

    const int tid  = threadIdx.x;
    const int wave = tid >> 5;
    const int lane = tid & 31;
    const int m    = lane & 15;
    const int hi   = lane >> 4;
    const int tok0 = blockIdx.x * 128 + wave * 16;
    const int d0   = blockIdx.y * 64;

    // Stage lora_B[0:64, d0:d0+64] transposed to f16 (once; K total = 64)
    #pragma unroll 4
    for (int i = 0; i < (DIM_R * 64) / 256; ++i) { // 16 iters
        int idx = tid + 256 * i;
        int k   = idx >> 6;
        int dd  = idx & 63;
        Bt[dd * LDSB_STRIDE + k] = (_Float16)Bw[(size_t)k * DIM_D + d0 + dd];
    }
    __syncthreads();

    v8f c[4];
    c[0] = (v8f){}; c[1] = (v8f){}; c[2] = (v8f){}; c[3] = (v8f){};

    const _Float16* __restrict__ Trow = T + (size_t)(tok0 + m) * DIM_R;

    #pragma unroll
    for (int ks = 0; ks < DIM_R; ks += 32) {
        // A fragment from row-major f16 T
        const v8h a0 = *(const v8h*)(Trow + ks + 8 * hi);
        const v8h a1 = *(const v8h*)(Trow + ks + 16 + 8 * hi);
        v16h a;
        #pragma unroll
        for (int e = 0; e < 8; ++e) { a[e] = a0[e]; a[8 + e] = a1[e]; }

        #pragma unroll
        for (int nb = 0; nb < 4; ++nb) {
            const v8h* bp = (const v8h*)&Bt[(nb * 16 + m) * LDSB_STRIDE + ks + 8 * hi];
            const v8h b0 = bp[0];
            const v8h b1 = bp[2];
            v16h b;
            #pragma unroll
            for (int e = 0; e < 8; ++e) { b[e] = b0[e]; b[8 + e] = b1[e]; }
            c[nb] = wmma_f16(a, b, c[nb]);
        }
    }

    // Epilogue: fold type_weight * LORA_SCALING (=2.0) into the store
    #pragma unroll
    for (int v = 0; v < 8; ++v) {
        int row = tok0 + v + 8 * hi;
        float w = tw[row] * 2.0f;
        #pragma unroll
        for (int nb = 0; nb < 4; ++nb)
            out[(size_t)row * DIM_D + d0 + nb * 16 + m] = w * c[nb][v];
    }
}

extern "C" void kernel_launch(void* const* d_in, const int* in_sizes, int n_in,
                              void* d_out, int out_size, void* d_ws, size_t ws_size,
                              hipStream_t stream) {
    const float* x      = (const float*)d_in[0];  // (8,2048,4096)
    const float* tw     = (const float*)d_in[1];  // (8,2048)
    const float* lora_A = (const float*)d_in[2];  // (4096,64)
    const float* lora_B = (const float*)d_in[3];  // (64,4096)
    float* out          = (float*)d_out;          // (8,2048,4096)
    _Float16* T         = (_Float16*)d_ws;        // 16384*64 f16 = 2 MB scratch

    lora_gemm1<<<dim3(N_TOK / 64), dim3(128), 0, stream>>>(x, lora_A, T);
    lora_gemm2<<<dim3(N_TOK / 128, DIM_D / 64), dim3(256), 0, stream>>>(T, lora_B, tw, out);
}
